// SSM_9680856285217
// MI455X (gfx1250) — matrix operand order repaired and checked by reading, not compiled, not run on hardware
//
#include <hip/hip_runtime.h>
#include <math.h>
#include <stdint.h>

// Problem dims (fixed by the reference)
#define L_    6
#define B_    4096
#define H_    2048
#define DIN_  512
#define DOUT_ 512
#define R_    4

// GEMM tiling
#define BM 128
#define BN 128
#define BK 64
#define LDA 72   // padded LDS row stride (elements) to avoid bank conflicts
#define LDB 72

typedef __bf16 bf16t;
typedef bf16t v16bf __attribute__((ext_vector_type(16)));
typedef float v8f   __attribute__((ext_vector_type(8)));
typedef unsigned int u32x4 __attribute__((ext_vector_type(4)));
typedef int v4i_t __attribute__((ext_vector_type(4)));

typedef v4i_t __attribute__((address_space(1))) as1_v4i;
typedef v4i_t __attribute__((address_space(3))) as3_v4i;

#if defined(__gfx1250__) && __has_builtin(__builtin_amdgcn_global_load_async_to_lds_b128)
#define USE_ASYNC 1
#else
#define USE_ASYNC 0
#endif

union FragU {
    v16bf bf;
    u32x4 u4[2];
};

__device__ __forceinline__ unsigned short f2bf(float f) {
    unsigned int u = __float_as_uint(f);
    u += 0x7FFFu + ((u >> 16) & 1u);   // round-to-nearest-even
    return (unsigned short)(u >> 16);
}

__device__ __forceinline__ float gelu_tanh(float x) {
    const float c0 = 0.7978845608028654f;  // sqrt(2/pi)
    const float c1 = 0.044715f;
    float t = tanhf(c0 * (x + c1 * x * x * x));
    return 0.5f * x * (1.0f + t);
}

__device__ __forceinline__ void wait_async0() {
#if USE_ASYNC
#if __has_builtin(__builtin_amdgcn_s_wait_asynccnt)
    __builtin_amdgcn_s_wait_asynccnt(0);
#else
    asm volatile("s_wait_asynccnt 0x0" ::: "memory");
#endif
#endif
}

#if USE_ASYNC
__device__ __forceinline__ void async_copy16(const unsigned short* g, unsigned short* l) {
    // Generic LDS pointer: low 32 bits are the LDS offset (flat aperture rule),
    // so integer truncation is the correct generic->AS3 conversion.
    __builtin_amdgcn_global_load_async_to_lds_b128(
        (as1_v4i*)(uintptr_t)g,
        (as3_v4i*)(unsigned int)(uintptr_t)l,
        0, 0);
}
#endif

// ---------------------------------------------------------------------------
// fp32 -> bf16 elementwise convert
// ---------------------------------------------------------------------------
__global__ __launch_bounds__(256) void f32_to_bf16_kernel(
        const float* __restrict__ in, unsigned short* __restrict__ out, int n) {
    int i = blockIdx.x * 256 + threadIdx.x;
    int stride = gridDim.x * 256;
    for (; i < n; i += stride) out[i] = f2bf(in[i]);
}

// ---------------------------------------------------------------------------
// Transpose-convert b_mat[l] : fp32 [K][N] -> bf16 [N][K]  (LDS 32x32 tiles)
// ---------------------------------------------------------------------------
__global__ __launch_bounds__(256) void trans_bf16_kernel(
        const float* __restrict__ bmat, unsigned short* __restrict__ bmT) {
    __shared__ float tile[32][33];
    const int l = blockIdx.z;
    const float* in = bmat + (size_t)l * H_ * H_;
    unsigned short* out = bmT + (size_t)l * H_ * H_;
    const int tx = threadIdx.x;          // 0..31
    const int ty = threadIdx.y;          // 0..7
    const int n0 = blockIdx.x * 32;
    const int k0 = blockIdx.y * 32;
#pragma unroll
    for (int j = 0; j < 4; ++j)
        tile[ty + 8 * j][tx] = in[(size_t)(k0 + ty + 8 * j) * H_ + n0 + tx];
    __syncthreads();
#pragma unroll
    for (int j = 0; j < 4; ++j)
        out[(size_t)(n0 + ty + 8 * j) * H_ + k0 + tx] = f2bf(tile[tx][ty + 8 * j]);
}

// ---------------------------------------------------------------------------
// s[b][r] = sum_h hs[b][h] * q[h][r]    (R = 4)
// ---------------------------------------------------------------------------
__global__ __launch_bounds__(256) void rank_kernel(
        const float* __restrict__ hsl, const float* __restrict__ ql,
        float* __restrict__ s) {
    __shared__ float red[256 * 4];
    const int b = blockIdx.x;
    const int tid = threadIdx.x;
    const float* hrow = hsl + (size_t)b * H_;
    float a0 = 0.f, a1 = 0.f, a2 = 0.f, a3 = 0.f;
    for (int h = tid; h < H_; h += 256) {
        float hv = hrow[h];
        float4 qv = *(const float4*)(ql + (size_t)h * 4);
        a0 += hv * qv.x; a1 += hv * qv.y; a2 += hv * qv.z; a3 += hv * qv.w;
    }
    red[tid * 4 + 0] = a0; red[tid * 4 + 1] = a1;
    red[tid * 4 + 2] = a2; red[tid * 4 + 3] = a3;
    for (int ofs = 128; ofs > 0; ofs >>= 1) {
        __syncthreads();
        if (tid < ofs) {
#pragma unroll
            for (int r = 0; r < 4; ++r)
                red[tid * 4 + r] += red[(tid + ofs) * 4 + r];
        }
    }
    if (tid == 0)
        *(float4*)(s + (size_t)b * 4) = make_float4(red[0], red[1], red[2], red[3]);
}

// ---------------------------------------------------------------------------
// Tiled bf16 WMMA GEMM: C[M][N] = A[M][K] * W[N][K]^T  (fp32 accumulate)
// Double-buffered LDS; global->LDS via async-to-LDS when available.
// Epilogue modes:
//   0: out_b = bf16(C + bias)                          (in-projection)
//   1: h = hs_in*adiag + s.p + C ; out_f = h ; out_b = bf16(gelu(h))  (layer)
//   2: out_f = C + bias                                (out-projection)
// Block: 256 threads = 8 waves; block tile 128x128; wave tile 32x64.
// ---------------------------------------------------------------------------
__global__ __launch_bounds__(256) void gemm_wmma_kernel(
        const unsigned short* __restrict__ A, int lda,
        const unsigned short* __restrict__ W, int ldw,
        int M, int N, int K, int mode,
        const float* __restrict__ bias,
        const float* __restrict__ hs_in,
        const float* __restrict__ adiag,
        const float* __restrict__ svec,
        const float* __restrict__ pmat,
        float* __restrict__ out_f,
        unsigned short* __restrict__ out_b) {
    __shared__ unsigned short lA[2][BM * LDA];
    __shared__ unsigned short lB[2][BN * LDB];

    const int tid   = threadIdx.x;
    const int mBase = blockIdx.y * BM;
    const int nBase = blockIdx.x * BN;
    const int wave  = tid >> 5;
    const int lane  = tid & 31;
    const int hl    = lane >> 4;    // 0 or 1 (lane half)
    const int n16   = lane & 15;
    const int wm    = (wave >> 1) * 32;   // wave row offset within block tile
    const int wn    = (wave & 1) * 64;    // wave col offset within block tile

    v8f acc[2][4];
#pragma unroll
    for (int mi = 0; mi < 2; ++mi)
#pragma unroll
        for (int ni = 0; ni < 4; ++ni)
#pragma unroll
            for (int r = 0; r < 8; ++r) acc[mi][ni][r] = 0.0f;

    // Stage one BK-slice of A and W tiles into LDS buffer bb.
    auto stage = [&](int k0, int bb) {
        for (int c = tid; c < (BM * BK / 8); c += 256) {   // 16B chunks
            int row = c >> 3;
            int off = (c & 7) * 8;
            const unsigned short* g = A + (size_t)(mBase + row) * lda + k0 + off;
            unsigned short* l = &lA[bb][row * LDA + off];
#if USE_ASYNC
            async_copy16(g, l);
#else
            *(u32x4*)l = *(const u32x4*)g;
#endif
        }
        for (int c = tid; c < (BN * BK / 8); c += 256) {
            int row = c >> 3;
            int off = (c & 7) * 8;
            const unsigned short* g = W + (size_t)(nBase + row) * ldw + k0 + off;
            unsigned short* l = &lB[bb][row * LDB + off];
#if USE_ASYNC
            async_copy16(g, l);
#else
            *(u32x4*)l = *(const u32x4*)g;
#endif
        }
    };

    const int nk = K / BK;
    stage(0, 0);
    wait_async0();
    __syncthreads();

    for (int kt = 0; kt < nk; ++kt) {
        const int bb = kt & 1;
        if (kt + 1 < nk) stage((kt + 1) * BK, bb ^ 1);   // overlap next copy with math

#pragma unroll
        for (int ks = 0; ks < BK; ks += 32) {
            FragU af[2], bf[4];
            // A fragment: lane (m = n16, half = hl); runs at K = ks+hl*8 and ks+16+hl*8
#pragma unroll
            for (int mi = 0; mi < 2; ++mi) {
                const unsigned short* base = &lA[bb][(wm + mi * 16 + n16) * LDA + ks + hl * 8];
                af[mi].u4[0] = *(const u32x4*)(base);
                af[mi].u4[1] = *(const u32x4*)(base + 16);
            }
            // B fragment: lane (n = n16, half = hl); runs at K = ks+hl*8 and ks+16+hl*8
#pragma unroll
            for (int ni = 0; ni < 4; ++ni) {
                const unsigned short* base = &lB[bb][(wn + ni * 16 + n16) * LDB + ks + hl * 8];
                bf[ni].u4[0] = *(const u32x4*)(base);
                bf[ni].u4[1] = *(const u32x4*)(base + 16);
            }
#pragma unroll
            for (int mi = 0; mi < 2; ++mi)
#pragma unroll
                for (int ni = 0; ni < 4; ++ni)
                    acc[mi][ni] = __builtin_amdgcn_wmma_f32_16x16x32_bf16(
                        false, af[mi].bf, false, bf[ni].bf,
                        (short)0, acc[mi][ni], false, false);
        }

        wait_async0();      // next-buffer copies complete (no-op on fallback path)
        __syncthreads();    // all waves done with buffer bb; buffer bb^1 visible
    }

    // ---------------- Epilogue ----------------
#pragma unroll
    for (int mi = 0; mi < 2; ++mi) {
#pragma unroll
        for (int ni = 0; ni < 4; ++ni) {
            const int gn = nBase + wn + ni * 16 + n16;
            const int rowBase = mBase + wm + mi * 16 + hl * 8;
            float ad = 0.f, bv = 0.f;
            float4 pv = make_float4(0.f, 0.f, 0.f, 0.f);
            if (mode == 1) {
                ad = adiag[gn];
                pv = *(const float4*)(pmat + (size_t)gn * 4);
            } else {
                bv = bias[gn];
            }
#pragma unroll
            for (int r = 0; r < 8; ++r) {
                const int gm = rowBase + r;
                float c = acc[mi][ni][r];
                if (mode == 0) {
                    out_b[(size_t)gm * N + gn] = f2bf(c + bv);
                } else if (mode == 2) {
                    out_f[(size_t)gm * N + gn] = c + bv;
                } else {
                    float4 sv = *(const float4*)(svec + (size_t)gm * 4);
                    float h = hs_in[(size_t)gm * N + gn] * ad +
                              sv.x * pv.x + sv.y * pv.y + sv.z * pv.z + sv.w * pv.w + c;
                    out_f[(size_t)gm * N + gn] = h;
                    out_b[(size_t)gm * N + gn] = f2bf(gelu_tanh(h));
                }
            }
        }
    }
}

// ---------------------------------------------------------------------------
extern "C" void kernel_launch(void* const* d_in, const int* in_sizes, int n_in,
                              void* d_out, int out_size, void* d_ws, size_t ws_size,
                              hipStream_t stream) {
    const float* hs    = (const float*)d_in[0];
    const float* x     = (const float*)d_in[1];
    const float* w_in  = (const float*)d_in[2];
    const float* b_in  = (const float*)d_in[3];
    const float* adiag = (const float*)d_in[4];
    const float* p     = (const float*)d_in[5];
    const float* q     = (const float*)d_in[6];
    const float* bmat  = (const float*)d_in[7];
    const float* w_out = (const float*)d_in[8];
    const float* b_out = (const float*)d_in[9];
    float* out = (float*)d_out;
    (void)in_sizes; (void)n_in; (void)out_size; (void)ws_size;

    char* ws = (char*)d_ws;
    size_t off = 0;
    auto wsAlloc = [&](size_t bytes) -> char* {
        char* pp = ws + off;
        off += (bytes + 255) & ~(size_t)255;
        return pp;
    };
    unsigned short* xb0   = (unsigned short*)wsAlloc((size_t)B_ * H_ * 2);
    unsigned short* xb1   = (unsigned short*)wsAlloc((size_t)B_ * H_ * 2);
    unsigned short* wbin  = (unsigned short*)wsAlloc((size_t)H_ * DIN_ * 2);
    unsigned short* wbout = (unsigned short*)wsAlloc((size_t)DOUT_ * H_ * 2);
    unsigned short* bmT   = (unsigned short*)wsAlloc((size_t)L_ * H_ * H_ * 2);
    float*          svec  = (float*)wsAlloc((size_t)B_ * R_ * 4);

    // Stage bf16 operands
    f32_to_bf16_kernel<<<1024, 256, 0, stream>>>(x, xb0, B_ * DIN_);
    f32_to_bf16_kernel<<<1024, 256, 0, stream>>>(w_in, wbin, H_ * DIN_);
    f32_to_bf16_kernel<<<1024, 256, 0, stream>>>(w_out, wbout, DOUT_ * H_);
    trans_bf16_kernel<<<dim3(H_ / 32, H_ / 32, L_), dim3(32, 8), 0, stream>>>(bmat, bmT);

    // In-projection: x @ w_in^T + b_in -> bf16 activations (xb1)
    gemm_wmma_kernel<<<dim3(H_ / BN, B_ / BM), 256, 0, stream>>>(
        xb0, DIN_, wbin, DIN_, B_, H_, DIN_, 0,
        b_in, nullptr, nullptr, nullptr, nullptr, nullptr, xb1);

    const unsigned short* cur = xb1;
    unsigned short* nxt = xb0;
    for (int i = 0; i < L_; ++i) {
        rank_kernel<<<B_, 256, 0, stream>>>(
            hs + (size_t)i * B_ * H_, q + (size_t)i * H_ * R_, svec);
        gemm_wmma_kernel<<<dim3(H_ / BN, B_ / BM), 256, 0, stream>>>(
            cur, H_, bmT + (size_t)i * H_ * H_, H_, B_, H_, H_, 1,
            nullptr, hs + (size_t)i * B_ * H_, adiag + (size_t)i * H_,
            svec, p + (size_t)i * H_ * R_,
            out + (size_t)i * B_ * H_, nxt);
        const unsigned short* t = cur;
        cur = nxt;
        nxt = (unsigned short*)t;
    }

    // Out-projection: gelu(h_last) @ w_out^T + b_out -> y
    gemm_wmma_kernel<<<dim3(DOUT_ / BN, B_ / BM), 256, 0, stream>>>(
        cur, H_, wbout, H_, B_, DOUT_, H_, 2,
        b_out, nullptr, nullptr, nullptr, nullptr,
        out + (size_t)L_ * B_ * H_, nullptr);
}
